// InfoNCELoss_86371792322729
// MI455X (gfx1250) — matrix operand order repaired and checked by reading, not compiled, not run on hardware
//
#include <hip/hip_runtime.h>
#include <hip/hip_bf16.h>
#include <math.h>

#define BATCH 16
#define NROWS 1024
#define CDIM  128
#define KNEG  64
#define INV_T (1.0f / 0.07f)

typedef __attribute__((ext_vector_type(16))) _Float16 v16h;
typedef __attribute__((ext_vector_type(8)))  _Float16 v8h;
typedef __attribute__((ext_vector_type(4)))  _Float16 v4h;
typedef __attribute__((ext_vector_type(8)))  float    v8f;

// ---------------------------------------------------------------------------
// Kernel 1: L2-normalize q and k rows (C=128), write f16 into workspace.
// One wave32 per row: 4 floats per lane + cross-lane reduction.
// ---------------------------------------------------------------------------
__global__ __launch_bounds__(256) void nrm_kernel(
    const float* __restrict__ q, const float* __restrict__ k,
    _Float16* __restrict__ qn, _Float16* __restrict__ kn, int rows_per_src)
{
    const int row  = blockIdx.x * 8 + (threadIdx.x >> 5);
    const int lane = threadIdx.x & 31;

    const float* src;
    _Float16*    dst;
    if (row < rows_per_src) {
        src = q  + (size_t)row * CDIM;
        dst = qn + (size_t)row * CDIM;
    } else {
        const int r2 = row - rows_per_src;
        src = k  + (size_t)r2 * CDIM;
        dst = kn + (size_t)r2 * CDIM;
    }

    const float4 x = *(const float4*)(src + lane * 4);
    float s = x.x * x.x + x.y * x.y + x.z * x.z + x.w * x.w;
    // wave32 reduction
    #pragma unroll
    for (int off = 16; off > 0; off >>= 1)
        s += __shfl_xor(s, off, 32);

    const float scale = 1.0f / fmaxf(sqrtf(s), 1e-12f);
    v4h o;
    o[0] = (_Float16)(x.x * scale);
    o[1] = (_Float16)(x.y * scale);
    o[2] = (_Float16)(x.z * scale);
    o[3] = (_Float16)(x.w * scale);
    *(v4h*)(dst + lane * 4) = o;
}

// ---------------------------------------------------------------------------
// Kernel 2: per (batch, 16-query tile), dense S = Qtile x Kn^T via WMMA into
// LDS (16 x 1024 f32), then gather + masked logsumexp, deterministic partials.
// blockDim = 128 (4 waves); wave w computes N-tiles [16w, 16w+16).
// ---------------------------------------------------------------------------
__global__ __launch_bounds__(128) void sim_kernel(
    const _Float16* __restrict__ qn, const _Float16* __restrict__ kn,
    const int* __restrict__ pidx, const int* __restrict__ nidx,
    float* __restrict__ totals, float* __restrict__ counts)
{
    __shared__ float S[16 * NROWS];   // 64 KB

    const int tid  = threadIdx.x;
    const int lane = tid & 31;
    const int wave = tid >> 5;        // 0..3
    const int hl   = (lane >> 4) & 1; // lane half
    const int lrow = lane & 15;

    const int b    = blockIdx.x >> 6; // 64 tiles per batch
    const int tile = blockIdx.x & 63;
    const int n0   = tile * 16;

    // --- A fragments: 16x128 query tile, 4 K-chunks of 32 ---------------
    // ISA layout (16-bit A 16x32): lanes 0-15 hold K{0..7, 16..23},
    // lanes 16-31 hold K{8..15, 24..31}.
    const _Float16* qrow = qn + (size_t)(b * NROWS + n0 + lrow) * CDIM;
    v16h afrag[4];
    #pragma unroll
    for (int kc = 0; kc < 4; ++kc) {
        const _Float16* p = qrow + kc * 32 + hl * 8;
        v8h lo = *(const v8h*)(p);
        v8h hi = *(const v8h*)(p + 16);
        afrag[kc] = __builtin_shufflevector(lo, hi,
            0, 1, 2, 3, 4, 5, 6, 7, 8, 9, 10, 11, 12, 13, 14, 15);
    }

    // --- dense S tile: each wave does 16 N-tiles x 4 WMMAs --------------
    for (int t = 0; t < 16; ++t) {
        const int ntile = wave * 16 + t;
        // B layout (16-bit B 32x16): lane(0-15)=column n with K{0..7, 16..23};
        // lanes 16-31 same column set, K{8..15, 24..31}.
        const _Float16* krow =
            kn + (size_t)(b * NROWS + ntile * 16 + lrow) * CDIM;
        v8f acc = {};
        #pragma unroll
        for (int kc = 0; kc < 4; ++kc) {
            const _Float16* p = krow + kc * 32 + hl * 8;
            v8h lo = *(const v8h*)(p);
            v8h hi = *(const v8h*)(p + 16);
            v16h bfrag = __builtin_shufflevector(lo, hi,
                0, 1, 2, 3, 4, 5, 6, 7, 8, 9, 10, 11, 12, 13, 14, 15);
            acc = __builtin_amdgcn_wmma_f32_16x16x32_f16(
                false, afrag[kc], false, bfrag, (short)0, acc, false, false);
        }
        // D layout: VGPR r -> M = r + 8*hl, N = lrow
        const int col = ntile * 16 + lrow;
        #pragma unroll
        for (int r = 0; r < 8; ++r)
            S[(r + hl * 8) * NROWS + col] = acc[r];
    }
    __syncthreads();

    // --- gather + masked logsumexp (threads 0..15, one query each) ------
    float tot = 0.0f, cnt = 0.0f;
    if (tid < 16) {
        const int   n  = n0 + tid;
        const int   pi = pidx[b * NROWS + n];
        const bool  pv = (pi >= 0);
        const int   pc = pv ? (pi & (NROWS - 1)) : 0;
        const float pos = S[tid * NROWS + pc] * INV_T;
        const int* nbase = nidx + (size_t)(b * NROWS + n) * KNEG;

        float mx = pv ? pos : -INFINITY;
        for (int j = 0; j < KNEG; ++j) {
            const int ni = nbase[j];
            if (ni >= 0) {
                const float v = S[tid * NROWS + (ni & (NROWS - 1))] * INV_T;
                mx = fmaxf(mx, v);
            }
        }
        if (pv) {
            float se = __expf(pos - mx);
            for (int j = 0; j < KNEG; ++j) {
                const int ni = nbase[j];
                if (ni >= 0) {
                    const float v = S[tid * NROWS + (ni & (NROWS - 1))] * INV_T;
                    se += __expf(v - mx);
                }
            }
            tot = (mx + __logf(se)) - pos;   // logsumexp - pos_sim
            cnt = 1.0f;
        }
        totals[blockIdx.x * 16 + tid] = tot;
        counts[blockIdx.x * 16 + tid] = cnt;
    }
}

// ---------------------------------------------------------------------------
// Kernel 3: deterministic reduction of 16384 partials -> scalar loss.
// ---------------------------------------------------------------------------
__global__ __launch_bounds__(256) void fin_kernel(
    const float* __restrict__ totals, const float* __restrict__ counts,
    float* __restrict__ out)
{
    __shared__ float st[256];
    __shared__ float sc[256];
    const int t = threadIdx.x;
    float tt = 0.0f, cc = 0.0f;
    for (int i = t; i < BATCH * NROWS; i += 256) {
        tt += totals[i];
        cc += counts[i];
    }
    st[t] = tt; sc[t] = cc;
    __syncthreads();
    for (int s = 128; s > 0; s >>= 1) {
        if (t < s) { st[t] += st[t + s]; sc[t] += sc[t + s]; }
        __syncthreads();
    }
    if (t == 0) {
        const float c = sc[0];
        out[0] = (c > 0.0f) ? (st[0] / fmaxf(c, 1.0f)) : 0.0f;
    }
}

// ---------------------------------------------------------------------------
extern "C" void kernel_launch(void* const* d_in, const int* in_sizes, int n_in,
                              void* d_out, int out_size, void* d_ws, size_t ws_size,
                              hipStream_t stream) {
    const float* q  = (const float*)d_in[0];
    const float* k  = (const float*)d_in[1];
    const int*   pi = (const int*)d_in[2];
    const int*   ni = (const int*)d_in[3];
    float*       out = (float*)d_out;

    // workspace layout
    char* ws = (char*)d_ws;
    const size_t half_elems = (size_t)BATCH * NROWS * CDIM;     // 2,097,152
    _Float16* qn = (_Float16*)ws;                               // 4 MB
    _Float16* kn = (_Float16*)(ws + half_elems * sizeof(_Float16));  // 4 MB
    float* totals = (float*)(ws + 2 * half_elems * sizeof(_Float16));
    float* counts = totals + (size_t)BATCH * NROWS;             // 16384 each

    const int rows_per_src = BATCH * NROWS;                     // 16384
    // 2 * 16384 rows, 8 rows per 256-thread block -> 4096 blocks
    nrm_kernel<<<(2 * rows_per_src) / 8, 256, 0, stream>>>(q, k, qn, kn,
                                                           rows_per_src);
    // one block per (batch, 16-query tile): 16 * 64 = 1024 blocks
    sim_kernel<<<BATCH * (NROWS / 16), 128, 0, stream>>>(qn, kn, pi, ni,
                                                         totals, counts);
    fin_kernel<<<1, 256, 0, stream>>>(totals, counts, out);
}
